// SimplifiedSpeechModel_90142773609027
// MI455X (gfx1250) — compile-verified, matrix operand order checked by reading, not run on hardware
//
#include <hip/hip_runtime.h>
#include <hip/hip_bf16.h>
#include <math.h>

// Model dims (fixed by the reference)
#define B_   128
#define T_   2048
#define F_   80
#define NC_  10
#define H_   32
#define M_   (B_ * T_)     // 262144 rows through the feature extractor
#define EPS_ 1e-5f

typedef __attribute__((ext_vector_type(16))) _Float16 v16h;
typedef __attribute__((ext_vector_type(8)))  _Float16 v8h;   // one 16B ds_load_b128
typedef __attribute__((ext_vector_type(2)))  _Float16 h2;
typedef __attribute__((ext_vector_type(8)))  float    v8f;

// CDNA5 async global->LDS copy (ASYNCcnt-tracked, bypasses VGPRs).
// GV addressing: 64-bit global address in a VGPR pair, LDS byte offset in VDST.
__device__ __forceinline__ void async_copy_b128(unsigned int lds_off, const void* gaddr)
{
    asm volatile("global_load_async_to_lds_b128 %0, %1, off"
                 :: "v"(lds_off), "v"(gaddr) : "memory");
}
__device__ __forceinline__ void wait_asynccnt0()
{
    asm volatile("s_wait_asynccnt 0x0" ::: "memory");
}

// WMMA f16 16x16x32 fragment element map (ISA 7.12.2):
//   lane[3:0] = M (A) / N (B);  khalf = lane[4]
//   element e (0..15):  k = (e>=8?16:0) + khalf*8 + ((e>>1)&3)*2 + (e&1)
// "Fragment order" LDS/global layout: frag[tile][lane][e] -> 32 contiguous
// bytes per (tile,lane), so one fragment = 2x ds_load_b128, conflict-free.

// ---------------------------------------------------------------------------
// Fused GEMM: C[M,N] = act(A[M,K]) @ W^T + bias.
//  - per-K-channel BN scale/shift + ReLU applied while staging A into LDS
//    (written directly in WMMA fragment order)
//  - entire f16 weight matrix (already in fragment order in global memory)
//    async-copied into LDS once per block via global_load_async_to_lds_b128
//  - per-N-channel sum / sum-sq (next layer's BN stats) fused into epilogue
// 256 threads = 8 wave32; each wave owns a 16-row strip and walks N in pairs
// of 16-col tiles so each A fragment feeds two independent WMMAs.
// ---------------------------------------------------------------------------
__global__ __launch_bounds__(256)
void gemm_bn_relu_wmma(const float*    __restrict__ A,
                       const _Float16* __restrict__ Wh,      // [N*Kpad] f16, fragment order
                       const float*    __restrict__ bias,    // [N] or null
                       const float*    __restrict__ bnScale, // [K] or null => identity load
                       const float*    __restrict__ bnShift, // [K]
                       float*          __restrict__ C,       // [M, N] raw (pre-BN) output
                       float*          __restrict__ oSum,    // [N] or null
                       float*          __restrict__ oSumSq,  // [N]
                       int N, int K, int Kpad)
{
    extern __shared__ __align__(32) _Float16 smem[];
    _Float16* ldsA = smem;                       // 128*Kpad f16, fragment order
    _Float16* ldsB = smem + 128 * Kpad;          // N*Kpad f16, fragment order
    const int ktiles = Kpad >> 5;
    const int blockM = blockIdx.x * 128;

    // Kick off the async B copy first so it overlaps the A transform below.
    {
        const unsigned int ldsBoff = (unsigned int)(uintptr_t)ldsB;
        const int nchunks = (N * Kpad) >> 3;     // 8 f16 = 16B per chunk
        for (int idx = threadIdx.x; idx < nchunks; idx += 256) {
            async_copy_b128(ldsBoff + idx * 16, (const char*)Wh + (size_t)idx * 16);
        }
    }

    // Stage the 128 x Kpad activation tile as f16 in fragment order.
    // Iterate row-major over (row, k-pair) so global reads stay coalesced;
    // scatter 4-byte pairs to the swizzled LDS destination.
    {
        const int kp2 = Kpad >> 1;
        for (int idx = threadIdx.x; idx < 128 * kp2; idx += 256) {
            const int r  = idx / kp2;
            const int k  = (idx - r * kp2) * 2;
            float v0 = 0.f, v1 = 0.f;
            const float* arow = A + (size_t)(blockM + r) * K;
            if (k < K)     v0 = arow[k];
            if (k + 1 < K) v1 = arow[k + 1];
            if (bnScale) {
                if (k < K)     v0 = fmaxf(v0 * bnScale[k] + bnShift[k], 0.f);
                if (k + 1 < K) v1 = fmaxf(v1 * bnScale[k + 1] + bnShift[k + 1], 0.f);
            }
            // (r,k) -> fragment order offset
            const int mtile = r >> 4, lm = r & 15;
            const int ktile = k >> 5, kk = k & 31;
            const int grp = kk >> 4, rem = kk & 15;
            const int khalf = rem >> 3, within = (rem & 7) >> 1;
            const int lane = lm + 16 * khalf;
            const int ep   = grp * 4 + within;      // element pair 0..7
            h2 pr; pr.x = (_Float16)v0; pr.y = (_Float16)v1;
            *(h2*)&ldsA[((size_t)(mtile * ktiles + ktile) * 32 + lane) * 16 + ep * 2] = pr;
        }
    }
    wait_asynccnt0();          // this wave's async LDS writes are complete
    __syncthreads();           // ... and visible to all waves

    const int lane  = threadIdx.x & 31;
    const int wave  = threadIdx.x >> 5;
    const int lm    = lane & 15;
    const int khalf = lane >> 4;
    const int mrow  = wave * 16;                 // wave's row strip within block

    for (int nt = 0; nt < N; nt += 32) {
        const int n0 = nt + lm;
        const int n1 = nt + 16 + lm;
        const int n0t = nt >> 4, n1t = n0t + 1;
        v8f acc0 = {}, acc1 = {};
        for (int kt = 0; kt < ktiles; ++kt) {
            const _Float16* ab = &ldsA[((size_t)(wave * ktiles + kt) * 32 + lane) * 16];
            const _Float16* b0 = &ldsB[((size_t)(n0t  * ktiles + kt) * 32 + lane) * 16];
            const _Float16* b1 = &ldsB[((size_t)(n1t  * ktiles + kt) * 32 + lane) * 16];
            v8h a_lo = *(const v8h*)ab,        a_hi = *(const v8h*)(ab + 8);
            v8h b0lo = *(const v8h*)b0,        b0hi = *(const v8h*)(b0 + 8);
            v8h b1lo = *(const v8h*)b1,        b1hi = *(const v8h*)(b1 + 8);
            v16h af, bf0, bf1;
            #pragma unroll
            for (int e = 0; e < 8; ++e) {
                af[e]  = a_lo[e];  af[8 + e]  = a_hi[e];
                bf0[e] = b0lo[e];  bf0[8 + e] = b0hi[e];
                bf1[e] = b1lo[e];  bf1[8 + e] = b1hi[e];
            }
            acc0 = __builtin_amdgcn_wmma_f32_16x16x32_f16(false, af, false, bf0,
                                                          (short)0, acc0, false, false);
            acc1 = __builtin_amdgcn_wmma_f32_16x16x32_f16(false, af, false, bf1,
                                                          (short)0, acc1, false, false);
        }
        // Epilogue: add bias, store raw (pre-BN) output, fuse BN statistics.
        #pragma unroll
        for (int q = 0; q < 2; ++q) {
            const int n = q ? n1 : n0;
            const v8f& acc = q ? acc1 : acc0;
            const float bn = bias ? bias[n] : 0.f;
            float ps = 0.f, pss = 0.f;
            #pragma unroll
            for (int r = 0; r < 8; ++r) {
                const int m = blockM + mrow + khalf * 8 + r;  // D: M = r + 8*lane[4]
                const float v = acc[r] + bn;
                C[(size_t)m * N + n] = v;
                ps += v; pss += v * v;
            }
            if (oSum) {
                // lanes L and L+16 share the same column n
                ps  += __shfl_down(ps, 16);
                pss += __shfl_down(pss, 16);
                if (lane < 16) {
                    atomicAdd(&oSum[n], ps);
                    atomicAdd(&oSumSq[n], pss);
                }
            }
        }
    }
}

// ---------------------------------------------------------------------------
// Convert fp32 weight matrix [N,K] to f16 [N*Kpad] in WMMA fragment order
// (zero padded in K).  Output index space: (tile = ntile*ktiles+ktile,
// lane, e) -> element 16*(32*tile+lane)+e.
// ---------------------------------------------------------------------------
__global__ void cvt_w_f16_frag(const float* __restrict__ W, _Float16* __restrict__ Wh,
                               int N, int K, int Kpad)
{
    const int idx = blockIdx.x * blockDim.x + threadIdx.x;
    if (idx >= N * Kpad) return;
    const int e    = idx & 15;
    const int lane = (idx >> 4) & 31;
    const int tile = idx >> 9;
    const int ktiles = Kpad >> 5;
    const int ntile = tile / ktiles;
    const int ktile = tile - ntile * ktiles;
    const int n = ntile * 16 + (lane & 15);
    const int khalf = lane >> 4;
    const int k = ktile * 32 + ((e >> 3) ? 16 : 0) + khalf * 8 + ((e >> 1) & 3) * 2 + (e & 1);
    Wh[idx] = (k < K) ? (_Float16)W[(size_t)n * K + k] : (_Float16)0.f;
}

__global__ void zero_f32(float* __restrict__ p, int n)
{
    const int i = blockIdx.x * blockDim.x + threadIdx.x;
    if (i < n) p[i] = 0.f;
}

// Per-channel BN: scale = g*rsqrt(var+eps), shift = beta - mean*scale.
__global__ void bn_finalize(const float* __restrict__ sum, const float* __restrict__ sumsq,
                            const float* __restrict__ g, const float* __restrict__ be,
                            float* __restrict__ scale, float* __restrict__ shift,
                            int n, float invM)
{
    const int i = blockIdx.x * blockDim.x + threadIdx.x;
    if (i >= n) return;
    const float m = sum[i] * invM;
    const float v = sumsq[i] * invM - m * m;
    const float a = g[i] * rsqrtf(v + EPS_);
    scale[i] = a;
    shift[i] = be[i] - m * a;
}

// ---------------------------------------------------------------------------
// GRU recurrence: one wave32 per (batch, direction).  Lane j owns gate rows
// (r_j, z_j, n_j); Whh rows live in VGPRs (96 floats/lane), h is broadcast
// through LDS with b128 reads.  xp = feats@Wih^T + bih is precomputed by the
// WMMA GEMM above.  Writes into gru_out[:, :, dir*32 + j] (fwd|bwd concat).
// ---------------------------------------------------------------------------
__global__ __launch_bounds__(32)
void gru_dir_kernel(const float* __restrict__ xp,      // [B,T,96] for this direction
                    const float* __restrict__ Whh,     // [96,32]
                    const float* __restrict__ bhh,     // [96]
                    float*       __restrict__ gru_out, // [B,T,64]
                    int dir)
{
    __shared__ __align__(16) float hsh[H_];
    const int b = blockIdx.x;
    const int j = threadIdx.x;  // 0..31
    float wr[H_], wz[H_], wn[H_];
    const float* Wr = Whh + (size_t)j * H_;
    const float* Wz = Whh + (size_t)(H_ + j) * H_;
    const float* Wn = Whh + (size_t)(2 * H_ + j) * H_;
    #pragma unroll
    for (int k4 = 0; k4 < H_ / 4; ++k4) {
        float4 a = ((const float4*)Wr)[k4];
        float4 c = ((const float4*)Wz)[k4];
        float4 d = ((const float4*)Wn)[k4];
        wr[4*k4] = a.x; wr[4*k4+1] = a.y; wr[4*k4+2] = a.z; wr[4*k4+3] = a.w;
        wz[4*k4] = c.x; wz[4*k4+1] = c.y; wz[4*k4+2] = c.z; wz[4*k4+3] = c.w;
        wn[4*k4] = d.x; wn[4*k4+1] = d.y; wn[4*k4+2] = d.z; wn[4*k4+3] = d.w;
    }
    const float br = bhh[j], bz = bhh[H_ + j], bnb = bhh[2 * H_ + j];
    const float* xb = xp + (size_t)b * T_ * 96;
    float* ob = gru_out + (size_t)b * T_ * 64 + dir * H_ + j;
    float hj = 0.f;
    hsh[j] = 0.f;
    __syncthreads();
    for (int step = 0; step < T_; ++step) {
        const int t = dir ? (T_ - 1 - step) : step;
        float h[H_];
        #pragma unroll
        for (int k4 = 0; k4 < H_ / 4; ++k4) {
            float4 v = ((const float4*)hsh)[k4];
            h[4*k4] = v.x; h[4*k4+1] = v.y; h[4*k4+2] = v.z; h[4*k4+3] = v.w;
        }
        float gr = br, gz = bz, gn = bnb;
        #pragma unroll
        for (int k = 0; k < H_; ++k) {
            gr = fmaf(wr[k], h[k], gr);
            gz = fmaf(wz[k], h[k], gz);
            gn = fmaf(wn[k], h[k], gn);
        }
        const float* xt = xb + (size_t)t * 96;
        const float r = 1.f / (1.f + __expf(-(xt[j] + gr)));
        const float z = 1.f / (1.f + __expf(-(xt[H_ + j] + gz)));
        const float n = tanhf(xt[2 * H_ + j] + r * gn);
        hj = (1.f - z) * n + z * hj;
        ob[(size_t)t * 64] = hj;
        __syncthreads();   // everyone has consumed old h
        hsh[j] = hj;
        __syncthreads();   // new h visible
    }
}

// ---------------------------------------------------------------------------
// Attention scores: scores[m] = tanh(gru[m]@Wa1^T + ba1) @ Wa2^T + ba2.
// ---------------------------------------------------------------------------
__global__ __launch_bounds__(256)
void attention_scores(const float* __restrict__ gru,
                      const float* __restrict__ Wa1, const float* __restrict__ ba1,
                      const float* __restrict__ Wa2, const float* __restrict__ ba2,
                      float* __restrict__ scores)
{
    __shared__ float w1[32 * 64];
    __shared__ float w2[32], b1s[32];
    for (int i = threadIdx.x; i < 32 * 64; i += 256) w1[i] = Wa1[i];
    if (threadIdx.x < 32) {
        w2[threadIdx.x]  = Wa2[threadIdx.x];
        b1s[threadIdx.x] = ba1[threadIdx.x];
    }
    __syncthreads();
    const int m = blockIdx.x * 256 + threadIdx.x;
    const float* g = gru + (size_t)m * 64;
    float gg[64];
    #pragma unroll
    for (int c = 0; c < 64; ++c) gg[c] = g[c];
    float s = ba2[0];
    for (int jj = 0; jj < 32; ++jj) {
        float d = b1s[jj];
        #pragma unroll
        for (int c = 0; c < 64; ++c) d = fmaf(w1[jj * 64 + c], gg[c], d);
        s = fmaf(tanhf(d), w2[jj], s);
    }
    scores[m] = s;
}

// ---------------------------------------------------------------------------
// Softmax over T + attention-weighted pooling: one block per batch element.
// ---------------------------------------------------------------------------
__global__ __launch_bounds__(256)
void softmax_pool(const float* __restrict__ scores,
                  const float* __restrict__ gru,
                  float* __restrict__ pooled)
{
    __shared__ float red[8];
    __shared__ float accs[8 * 64];
    __shared__ float bval[2];
    const int b = blockIdx.x, tid = threadIdx.x;
    const int lane = tid & 31, wave = tid >> 5;
    const float* s = scores + (size_t)b * T_;

    float mx = -3.4e38f;
    for (int t = tid; t < T_; t += 256) mx = fmaxf(mx, s[t]);
    for (int o = 16; o > 0; o >>= 1) mx = fmaxf(mx, __shfl_down(mx, o));
    if (lane == 0) red[wave] = mx;
    __syncthreads();
    if (tid == 0) {
        float m = red[0];
        for (int w = 1; w < 8; ++w) m = fmaxf(m, red[w]);
        bval[0] = m;
    }
    __syncthreads();
    mx = bval[0];

    float se = 0.f;
    for (int t = tid; t < T_; t += 256) se += __expf(s[t] - mx);
    for (int o = 16; o > 0; o >>= 1) se += __shfl_down(se, o);
    if (lane == 0) red[wave] = se;
    __syncthreads();
    if (tid == 0) {
        float a = 0.f;
        for (int w = 0; w < 8; ++w) a += red[w];
        bval[1] = a;
    }
    __syncthreads();
    const float inv = 1.f / bval[1];

    float acc[64];
    #pragma unroll
    for (int c = 0; c < 64; ++c) acc[c] = 0.f;
    for (int t = tid; t < T_; t += 256) {
        const float w = __expf(s[t] - mx);
        const float* g = gru + ((size_t)b * T_ + t) * 64;
        #pragma unroll
        for (int c = 0; c < 64; ++c) acc[c] = fmaf(w, g[c], acc[c]);
    }
    #pragma unroll
    for (int c = 0; c < 64; ++c)
        for (int o = 16; o > 0; o >>= 1) acc[c] += __shfl_down(acc[c], o);
    if (lane == 0) {
        #pragma unroll
        for (int c = 0; c < 64; ++c) accs[wave * 64 + c] = acc[c];
    }
    __syncthreads();
    if (tid < 64) {
        float tot = 0.f;
        for (int w = 0; w < 8; ++w) tot += accs[w * 64 + tid];
        pooled[(size_t)b * 64 + tid] = tot * inv;
    }
}

// ---------------------------------------------------------------------------
// Classifier: Linear(64->32) + BN(batch of 128) + ReLU + Linear(32->10).
// Tiny: single 256-thread block, everything through LDS.
// ---------------------------------------------------------------------------
__global__ __launch_bounds__(256)
void classifier(const float* __restrict__ pooled,
                const float* __restrict__ Wc1, const float* __restrict__ bc1,
                const float* __restrict__ gc,  const float* __restrict__ bec,
                const float* __restrict__ Wc2, const float* __restrict__ bc2,
                float* __restrict__ out)
{
    __shared__ float c1[B_ * 32];
    __shared__ float aL[32], bL[32];
    for (int idx = threadIdx.x; idx < B_ * 32; idx += 256) {
        const int i = idx >> 5, j = idx & 31;
        float d = bc1[j];
        for (int k = 0; k < 64; ++k) d = fmaf(pooled[i * 64 + k], Wc1[j * 64 + k], d);
        c1[idx] = d;
    }
    __syncthreads();
    if (threadIdx.x < 32) {
        const int j = threadIdx.x;
        float m = 0.f;
        for (int i = 0; i < B_; ++i) m += c1[i * 32 + j];
        m *= (1.f / B_);
        float v = 0.f;
        for (int i = 0; i < B_; ++i) { const float d = c1[i * 32 + j] - m; v += d * d; }
        v *= (1.f / B_);
        const float a = gc[j] * rsqrtf(v + EPS_);
        aL[j] = a;
        bL[j] = bec[j] - m * a;
    }
    __syncthreads();
    for (int idx = threadIdx.x; idx < B_ * NC_; idx += 256) {
        const int i = idx / NC_, c = idx - i * NC_;
        float d = bc2[c];
        for (int j = 0; j < 32; ++j) {
            const float h = fmaxf(c1[i * 32 + j] * aL[j] + bL[j], 0.f);
            d = fmaf(h, Wc2[c * 32 + j], d);
        }
        out[idx] = d;
    }
}

// ---------------------------------------------------------------------------
extern "C" void kernel_launch(void* const* d_in, const int* in_sizes, int n_in,
                              void* d_out, int out_size, void* d_ws, size_t ws_size,
                              hipStream_t stream)
{
    (void)in_sizes; (void)n_in; (void)out_size; (void)ws_size;
    const float* x     = (const float*)d_in[0];
    const float* W1    = (const float*)d_in[1];
    const float* b1    = (const float*)d_in[2];
    const float* g1    = (const float*)d_in[3];
    const float* be1   = (const float*)d_in[4];
    const float* W2    = (const float*)d_in[5];
    const float* b2    = (const float*)d_in[6];
    const float* g2    = (const float*)d_in[7];
    const float* be2   = (const float*)d_in[8];
    const float* W3    = (const float*)d_in[9];
    const float* b3    = (const float*)d_in[10];
    const float* g3    = (const float*)d_in[11];
    const float* be3   = (const float*)d_in[12];
    const float* Wih_f = (const float*)d_in[13];
    const float* Whh_f = (const float*)d_in[14];
    const float* bih_f = (const float*)d_in[15];
    const float* bhh_f = (const float*)d_in[16];
    const float* Wih_b = (const float*)d_in[17];
    const float* Whh_b = (const float*)d_in[18];
    const float* bih_b = (const float*)d_in[19];
    const float* bhh_b = (const float*)d_in[20];
    const float* Wa1   = (const float*)d_in[21];
    const float* ba1   = (const float*)d_in[22];
    const float* Wa2   = (const float*)d_in[23];
    const float* ba2   = (const float*)d_in[24];
    const float* Wc1   = (const float*)d_in[25];
    const float* bc1   = (const float*)d_in[26];
    const float* gc    = (const float*)d_in[27];
    const float* bec   = (const float*)d_in[28];
    const float* Wc2   = (const float*)d_in[29];
    const float* bc2   = (const float*)d_in[30];

    // ---- workspace carve-out (256-byte aligned regions) ----
    char* ws = (char*)d_ws;
    size_t off = 0;
    auto carve = [&](size_t bytes) -> char* {
        char* p = ws + off;
        off += (bytes + 255) & ~(size_t)255;
        return p;
    };
    _Float16* Wh1  = (_Float16*)carve((size_t)256 * 96  * 2);
    _Float16* Wh2  = (_Float16*)carve((size_t)128 * 256 * 2);
    _Float16* Wh3  = (_Float16*)carve((size_t)64  * 128 * 2);
    _Float16* Whf  = (_Float16*)carve((size_t)96  * 64  * 2);
    _Float16* Whb  = (_Float16*)carve((size_t)96  * 64  * 2);
    float* stats   = (float*)carve(896 * 4);     // sums/sumsqs: 256+256+128+128+64+64
    float* sum1 = stats, *sq1 = stats + 256;
    float* sum2 = stats + 512, *sq2 = stats + 640;
    float* sum3 = stats + 768, *sq3 = stats + 832;
    float* sc   = (float*)carve(896 * 4);        // scale/shift per layer
    float* s1v = sc, *h1v = sc + 256;
    float* s2v = sc + 512, *h2v = sc + 640;
    float* s3v = sc + 768, *h3v = sc + 832;
    float* regA   = (float*)carve((size_t)M_ * 256 * 4); // H1, later xp_f|xp_b
    float* regB   = (float*)carve((size_t)M_ * 128 * 4); // H2, later gru_out|scores
    float* H3     = (float*)carve((size_t)M_ * 64 * 4);
    float* pooled = (float*)carve((size_t)B_ * 64 * 4);

    float* H1     = regA;
    float* xp_f   = regA;
    float* xp_b   = regA + (size_t)M_ * 96;
    float* H2     = regB;
    float* gru    = regB;
    float* scores = regB + (size_t)M_ * 64;

    const float invM = 1.f / (float)M_;
    // LDS = A tile (128*Kpad f16, fragment order) + weights (N*Kpad f16).
    auto shmem = [](int N, int Kpad) {
        return (size_t)(128 + N) * Kpad * sizeof(_Float16);
    };

    // ---- pipeline ----
    zero_f32<<<4, 256, 0, stream>>>(stats, 896);
    cvt_w_f16_frag<<<(256 * 96 + 255) / 256, 256, 0, stream>>>(W1, Wh1, 256, 80, 96);
    cvt_w_f16_frag<<<(128 * 256 + 255) / 256, 256, 0, stream>>>(W2, Wh2, 128, 256, 256);
    cvt_w_f16_frag<<<(64 * 128 + 255) / 256, 256, 0, stream>>>(W3, Wh3, 64, 128, 128);
    cvt_w_f16_frag<<<(96 * 64 + 255) / 256, 256, 0, stream>>>(Wih_f, Whf, 96, 64, 64);
    cvt_w_f16_frag<<<(96 * 64 + 255) / 256, 256, 0, stream>>>(Wih_b, Whb, 96, 64, 64);

    // Layer 1: x -> H1 (raw), stats for BN1
    gemm_bn_relu_wmma<<<M_ / 128, 256, shmem(256, 96), stream>>>(
        x, Wh1, b1, nullptr, nullptr, H1, sum1, sq1, 256, 80, 96);
    bn_finalize<<<1, 256, 0, stream>>>(sum1, sq1, g1, be1, s1v, h1v, 256, invM);

    // Layer 2: BN1(H1)+ReLU -> H2 (raw), stats for BN2
    gemm_bn_relu_wmma<<<M_ / 128, 256, shmem(128, 256), stream>>>(
        H1, Wh2, b2, s1v, h1v, H2, sum2, sq2, 128, 256, 256);
    bn_finalize<<<1, 256, 0, stream>>>(sum2, sq2, g2, be2, s2v, h2v, 128, invM);

    // Layer 3: BN2(H2)+ReLU -> H3 (raw), stats for BN3
    gemm_bn_relu_wmma<<<M_ / 128, 256, shmem(64, 128), stream>>>(
        H2, Wh3, b3, s2v, h2v, H3, sum3, sq3, 64, 128, 128);
    bn_finalize<<<1, 256, 0, stream>>>(sum3, sq3, g3, be3, s3v, h3v, 64, invM);

    // GRU input projections: BN3(H3)+ReLU @ Wih^T + bih  (fwd & bwd)
    gemm_bn_relu_wmma<<<M_ / 128, 256, shmem(96, 64), stream>>>(
        H3, Whf, bih_f, s3v, h3v, xp_f, nullptr, nullptr, 96, 64, 64);
    gemm_bn_relu_wmma<<<M_ / 128, 256, shmem(96, 64), stream>>>(
        H3, Whb, bih_b, s3v, h3v, xp_b, nullptr, nullptr, 96, 64, 64);

    // Recurrences: one wave32 per (batch, direction)
    gru_dir_kernel<<<B_, 32, 0, stream>>>(xp_f, Whh_f, bhh_f, gru, 0);
    gru_dir_kernel<<<B_, 32, 0, stream>>>(xp_b, Whh_b, bhh_b, gru, 1);

    // Attention + pooling + classifier
    attention_scores<<<M_ / 256, 256, 0, stream>>>(gru, Wa1, ba1, Wa2, ba2, scores);
    softmax_pool<<<B_, 256, 0, stream>>>(scores, gru, pooled);
    classifier<<<1, 256, 0, stream>>>(pooled, Wc1, bc1, gc, bec, Wc2, bc2,
                                      (float*)d_out);
}
